// VQVAE_61873298866728
// MI455X (gfx1250) — matrix operand order repaired and checked by reading, not compiled, not run on hardware
//
#include <hip/hip_runtime.h>

typedef __attribute__((ext_vector_type(16))) __bf16 v16bf;
typedef __attribute__((ext_vector_type(8)))  float  v8f;
typedef unsigned short u16;

#define N_ROWS 32768   // 8*64*64
#define DIM    256
#define KCODES 8192
#define NTILES (KCODES / 16)
#define COMMIT_COST 0.25f

// ---------------------------------------------------------------------------
// Prep: split codebook fp32 -> bf16 hi/lo, compute per-code squared norms.
// One block (256 threads) per codebook row.
// ---------------------------------------------------------------------------
__global__ __launch_bounds__(256)
void vq_prep(const float* __restrict__ cb,
             u16* __restrict__ cb_hi, u16* __restrict__ cb_lo,
             float* __restrict__ cnorm) {
    const int k = blockIdx.x;
    const int t = threadIdx.x;
    const float x = cb[(size_t)k * DIM + t];
    const __bf16 h = (__bf16)x;
    const __bf16 l = (__bf16)(x - (float)h);
    cb_hi[(size_t)k * DIM + t] = __builtin_bit_cast(u16, h);
    cb_lo[(size_t)k * DIM + t] = __builtin_bit_cast(u16, l);

    float s = x * x;
#pragma unroll
    for (int m = 16; m >= 1; m >>= 1) s += __shfl_xor(s, m, 32);
    __shared__ float red[8];
    if ((t & 31) == 0) red[t >> 5] = s;
    __syncthreads();
    if (t == 0) {
        float tot = 0.f;
#pragma unroll
        for (int i = 0; i < 8; ++i) tot += red[i];
        cnorm[k] = tot;
    }
}

// ---------------------------------------------------------------------------
// Fused: score = ||c||^2 - 2 z.c via bf16x3 WMMA (3 independent accumulator
// chains + double-buffered B-fragment registers), running argmin over K,
// gather winning code rows, straight-through output, commitment loss.
// Block = 256 threads (8 waves); each wave owns 16 z-rows (A resident in
// VGPRs); block shares a double-buffered 16-code B tile in LDS (one barrier
// per iteration, staging overlaps WMMA).
// ---------------------------------------------------------------------------
__global__ __launch_bounds__(256)
void vq_main(const float* __restrict__ z, const float* __restrict__ cb,
             const u16* __restrict__ cb_hi, const u16* __restrict__ cb_lo,
             const float* __restrict__ cnorm,
             float* __restrict__ zq, float* __restrict__ loss,
             float* __restrict__ idx_out) {
    __shared__ __align__(32) u16 ldsB[2][2][16 * DIM];  // [buf][hi/lo][col*DIM+d], 32 KB
    __shared__ int   lds_idx[128];
    __shared__ float lds_loss;

    const int t       = threadIdx.x;
    const int lane    = t & 31;
    const int wave    = t >> 5;
    const int halfsel = lane >> 4;     // 0: lanes 0-15, 1: lanes 16-31
    const int lcol    = lane & 15;     // this lane's N-column slot
    const int rowBase = blockIdx.x * 128 + wave * 16;

    if (t == 0) lds_loss = 0.f;

    // ---- Build resident A fragments: bf16 hi/lo of (-2 * z_row) ----------
    // A 16x32 bf16 layout: lane row = lcol; elements 0..15 = K d0+halfsel*16..+15
    // (contiguous, matching B).
    v16bf a_hi[8], a_lo[8];
    {
        const float* zrow = z + (size_t)(rowBase + lcol) * DIM;
#pragma unroll
        for (int c = 0; c < 8; ++c) {
            const int d0 = c * 32;
            const int gA = d0 + halfsel * 16;
            const int gB = d0 + halfsel * 16 + 8;
            float buf[16];
            *(float4*)(buf + 0)  = *(const float4*)(zrow + gA);
            *(float4*)(buf + 4)  = *(const float4*)(zrow + gA + 4);
            *(float4*)(buf + 8)  = *(const float4*)(zrow + gB);
            *(float4*)(buf + 12) = *(const float4*)(zrow + gB + 4);
#pragma unroll
            for (int e = 0; e < 16; ++e) {
                const float x = -2.0f * buf[e];
                const __bf16 h = (__bf16)x;
                a_hi[c][e] = h;
                a_lo[c][e] = (__bf16)(x - (float)h);
            }
        }
    }

    float best[8];
    int   bidx[8];
#pragma unroll
    for (int r = 0; r < 8; ++r) { best[r] = 3.4e38f; bidx[r] = 0; }

    // ---- Stage tile 0 into buffer 0 --------------------------------------
    {
        const uint4* srcH = (const uint4*)(cb_hi);
        const uint4* srcL = (const uint4*)(cb_lo);
        uint4* dstH = (uint4*)&ldsB[0][0][0];
        uint4* dstL = (uint4*)&ldsB[0][1][0];
        dstH[t]       = srcH[t];
        dstH[t + 256] = srcH[t + 256];
        dstL[t]       = srcL[t];
        dstL[t + 256] = srcL[t + 256];
    }

    // ---- Sweep the codebook, 16 codes per iteration, double-buffered ------
    for (int i = 0; i < NTILES; ++i) {
        const int n   = i * 16;
        const int buf = i & 1;
        // Barrier: tile i fully staged, and buffer buf^1 fully consumed.
        __syncthreads();

        if (i + 1 < NTILES) {  // stage next tile; overlaps with WMMA below
            const uint4* srcH = (const uint4*)(cb_hi + (size_t)(n + 16) * DIM);
            const uint4* srcL = (const uint4*)(cb_lo + (size_t)(n + 16) * DIM);
            uint4* dstH = (uint4*)&ldsB[buf ^ 1][0][0];
            uint4* dstL = (uint4*)&ldsB[buf ^ 1][1][0];
            dstH[t]       = srcH[t];
            dstH[t + 256] = srcH[t + 256];
            dstL[t]       = srcL[t];
            dstL[t + 256] = srcL[t + 256];
        }
        if (i + 2 < NTILES && t < 128) {  // prefetch tile i+2 -> global_prefetch_b8
            __builtin_prefetch(cb_hi + (size_t)(n + 32) * DIM + (size_t)t * 32, 0, 1);
            __builtin_prefetch(cb_lo + (size_t)(n + 32) * DIM + (size_t)t * 32, 0, 1);
        }

        const float cn = cnorm[n + lcol];

        // B 32x16 bf16 layout: lane col = lcol, 16 contiguous K-values
        // starting at d0 + halfsel*16 -> 32-byte contiguous LDS read.
        const u16* bbaseH = &ldsB[buf][0][lcol * DIM + halfsel * 16];
        const u16* bbaseL = &ldsB[buf][1][lcol * DIM + halfsel * 16];

        // Three independent WMMA accumulation chains (hi*hi, hi*lo, lo*hi)
        // with explicit double-buffered B-fragment registers: group c+1's
        // ds_loads issue before group c's WMMAs, so waits are partial
        // instead of draining DScnt to zero every chunk.
        v8f acc0 = {}, acc1 = {}, acc2 = {};
        v16bf bh[2], bl[2];
        bh[0] = *(const v16bf*)(bbaseH);
        bl[0] = *(const v16bf*)(bbaseL);
#pragma unroll
        for (int c = 0; c < 8; ++c) {
            const int cur = c & 1;
            const int nxt = cur ^ 1;
            if (c + 1 < 8) {
                bh[nxt] = *(const v16bf*)(bbaseH + (c + 1) * 32);
                bl[nxt] = *(const v16bf*)(bbaseL + (c + 1) * 32);
            }
            acc0 = __builtin_amdgcn_wmma_f32_16x16x32_bf16(false, a_hi[c], false, bh[cur],
                                                           (short)0, acc0, false, false);
            acc1 = __builtin_amdgcn_wmma_f32_16x16x32_bf16(false, a_hi[c], false, bl[cur],
                                                           (short)0, acc1, false, false);
            acc2 = __builtin_amdgcn_wmma_f32_16x16x32_bf16(false, a_lo[c], false, bh[cur],
                                                           (short)0, acc2, false, false);
        }

        const int col = n + lcol;
#pragma unroll
        for (int r = 0; r < 8; ++r) {
            const float s = acc0[r] + acc1[r] + acc2[r] + cn;
            if (s < best[r]) { best[r] = s; bidx[r] = col; }
        }
    }

    // ---- Cross-lane argmin per row (ties -> smaller index, like argmin) ---
    // C layout: VGPR r holds row r (lanes 0-15) and row r+8 (lanes 16-31);
    // xor masks 1,2,4,8 stay within each 16-lane half.
#pragma unroll
    for (int r = 0; r < 8; ++r) {
        float s = best[r];
        int   i = bidx[r];
#pragma unroll
        for (int m = 8; m >= 1; m >>= 1) {
            const float so = __shfl_xor(s, m, 32);
            const int   io = __shfl_xor(i, m, 32);
            if (so < s || (so == s && io < i)) { s = so; i = io; }
        }
        if (lcol == 0) {
            const int mloc = r + halfsel * 8;
            lds_idx[wave * 16 + mloc] = i;
            idx_out[rowBase + mloc] = (float)i;
        }
    }
    __syncthreads();

    // ---- Gather winning codes, straight-through output, loss --------------
    float lsum = 0.f;
    for (int m = 0; m < 16; ++m) {
        const int code = lds_idx[wave * 16 + m];
        const int grow = rowBase + m;
        const float* crow = cb + (size_t)code * DIM;
        const float* zr   = z  + (size_t)grow * DIM;
        float*       qr   = zq + (size_t)grow * DIM;
#pragma unroll
        for (int e = 0; e < 8; ++e) {
            const int d = lane + e * 32;
            const float cv = crow[d];
            const float zv = zr[d];
            qr[d] = zv + (cv - zv);      // straight-through, matches reference fp
            const float dif = zv - cv;
            lsum += dif * dif;
        }
    }
#pragma unroll
    for (int m = 16; m >= 1; m >>= 1) lsum += __shfl_xor(lsum, m, 32);
    if (lane == 0) atomicAdd(&lds_loss, lsum);
    __syncthreads();
    if (t == 0)
        atomicAdd(loss, lds_loss * (COMMIT_COST / ((float)N_ROWS * (float)DIM)));
}

// ---------------------------------------------------------------------------
extern "C" void kernel_launch(void* const* d_in, const int* in_sizes, int n_in,
                              void* d_out, int out_size, void* d_ws, size_t ws_size,
                              hipStream_t stream) {
    const float* z  = (const float*)d_in[0];   // (8,64,64,256) fp32
    const float* cb = (const float*)d_in[1];   // (8192,256) fp32

    // d_out layout: [z_q: N*D][loss: 1][indices(as float): N]
    float* out   = (float*)d_out;
    float* zq    = out;
    float* loss  = out + (size_t)N_ROWS * DIM;
    float* idxf  = loss + 1;

    // workspace: cb_hi (4MB) | cb_lo (4MB) | cnorm (32KB) = ~8.4 MB
    u16*   cb_hi = (u16*)d_ws;
    u16*   cb_lo = cb_hi + (size_t)KCODES * DIM;
    float* cnorm = (float*)(cb_lo + (size_t)KCODES * DIM);

    hipMemsetAsync(loss, 0, sizeof(float), stream);
    vq_prep<<<KCODES, 256, 0, stream>>>(cb, cb_hi, cb_lo, cnorm);
    vq_main<<<N_ROWS / 128, 256, 0, stream>>>(z, cb, cb_hi, cb_lo, cnorm,
                                              zq, loss, idxf);
}
